// VectorCollapseEngine_2705829396737
// MI455X (gfx1250) — matrix operand order repaired and checked by reading, not compiled, not run on hardware
//
#include <hip/hip_runtime.h>
#include <math.h>

// ---------------------------------------------------------------------------
// Types
// ---------------------------------------------------------------------------
typedef __bf16 bf16_t;
typedef __attribute__((ext_vector_type(16))) __bf16 v16bf;
typedef __attribute__((ext_vector_type(8)))  __bf16 bf16x8;
typedef __attribute__((ext_vector_type(8)))  float  v8f;
typedef __attribute__((ext_vector_type(4)))  float  f32x4;
typedef __attribute__((ext_vector_type(4)))  unsigned int u32x4;

union FragBF { v16bf v; u32x4 q[2]; };

#define DIM   2048
#define BATCH 16384

// ---------------------------------------------------------------------------
// CDNA5 async Global->LDS DMA (ASYNCcnt-tracked), 16 bytes per lane per issue.
// LDS operand is the hardware LDS address = low 32 bits of the generic pointer
// (addrspacecast generic->local truncates on amdgcn).
// ---------------------------------------------------------------------------
__device__ __forceinline__ void vce_async_b128(const bf16_t* g, bf16_t* l) {
    asm volatile("global_load_async_to_lds_b128 %0, %1, off"
                 :: "v"((unsigned)(size_t)l),
                    "v"((unsigned long long)(size_t)g)
                 : "memory");
}
__device__ __forceinline__ void vce_wait_async0() {
    asm volatile("s_wait_asynccnt 0x0" ::: "memory");
}
// copy 64 contiguous bytes (one K-tile row slice) as 4 async b128 transfers
__device__ __forceinline__ void vce_async_row64(const bf16_t* g, bf16_t* l) {
    #pragma unroll
    for (int i = 0; i < 4; ++i) vce_async_b128(g + 8 * i, l + 8 * i);
}

// ---------------------------------------------------------------------------
// Normalize the 3 anchor vectors -> dirs[3][DIM]  (single block)
// ---------------------------------------------------------------------------
__global__ void vce_prep_dirs(const float* __restrict__ aE,
                              const float* __restrict__ aC,
                              const float* __restrict__ aN,
                              float* __restrict__ dirs) {
    __shared__ float red[8];
    const int tid  = threadIdx.x;
    const int lane = tid & 31, wave = tid >> 5;
    const float* src[3] = {aE, aC, aN};
    for (int v = 0; v < 3; ++v) {
        const float* a = src[v];
        float s = 0.f;
        for (int j = tid; j < DIM; j += 256) { float x = a[j]; s += x * x; }
        #pragma unroll
        for (int off = 16; off; off >>= 1) s += __shfl_down(s, off, 32);
        if (lane == 0) red[wave] = s;
        __syncthreads();
        float tot = 0.f;
        #pragma unroll
        for (int w = 0; w < 8; ++w) tot += red[w];
        const float inv = 1.f / fmaxf(sqrtf(tot), 1e-12f);
        for (int j = tid; j < DIM; j += 256) dirs[v * DIM + j] = a[j] * inv;
        __syncthreads();
    }
}

// ---------------------------------------------------------------------------
// W [N,K] f32  ->  Wt [K,N] bf16   (LDS-tiled transpose + convert)
// ---------------------------------------------------------------------------
__global__ __launch_bounds__(256) void vce_wcvt(const float* __restrict__ W,
                                                bf16_t* __restrict__ Wt) {
    __shared__ float t[32][33];
    const int tx = threadIdx.x & 31;
    const int ty = threadIdx.x >> 5;           // 0..7
    const int n0 = blockIdx.y * 32;
    const int k0 = blockIdx.x * 32;
    #pragma unroll
    for (int r = ty; r < 32; r += 8)
        t[r][tx] = W[(size_t)(n0 + r) * DIM + k0 + tx];
    __syncthreads();
    #pragma unroll
    for (int r = ty; r < 32; r += 8)
        Wt[(size_t)(k0 + r) * DIM + n0 + tx] = (bf16_t)t[tx][r];
}

// ---------------------------------------------------------------------------
// f32 -> bf16 elementwise (8 elems / thread, vectorized)
// ---------------------------------------------------------------------------
__global__ __launch_bounds__(256) void vce_cvt_h(const float* __restrict__ x,
                                                 bf16_t* __restrict__ y) {
    const size_t base = ((size_t)blockIdx.x * 256 + threadIdx.x) * 8;
    f32x4 a = *(const f32x4*)(x + base);
    f32x4 b = *(const f32x4*)(x + base + 4);
    bf16x8 o;
    o[0] = (bf16_t)a[0]; o[1] = (bf16_t)a[1]; o[2] = (bf16_t)a[2]; o[3] = (bf16_t)a[3];
    o[4] = (bf16_t)b[0]; o[5] = (bf16_t)b[1]; o[6] = (bf16_t)b[2]; o[7] = (bf16_t)b[3];
    *(bf16x8*)(y + base) = o;
}

// ---------------------------------------------------------------------------
// GEMM inner step: all 12 LDS fragment loads issued into distinct registers,
// then 8 WMMAs. LDA/LDB are padded bf16 strides (bank-conflict-free, 16B-aligned).
// ---------------------------------------------------------------------------
__device__ __forceinline__ void vce_kstep(const bf16_t* __restrict__ cA,
                                          const bf16_t* __restrict__ cB,
                                          v8f acc[4][2],
                                          int lane, int wm, int wn) {
    constexpr int LDA = 40;
    constexpr int LDB = 136;
    // B fragments: 32x16 (KxN), col = lane&15, halves split K {0..7,16..23}/{8..15,24..31}
    FragBF fb[2];
    #pragma unroll
    for (int j = 0; j < 2; ++j) {
        const bf16_t* pb = cB + ((lane >> 4) * 8) * LDB + wn + j * 16 + (lane & 15);
        #pragma unroll
        for (int s = 0; s < 16; ++s)
            fb[j].v[s] = pb[((s < 8) ? s : (8 + s)) * LDB];
    }
    // A fragments: 16x32 (MxK), row = lane&15, halves split K {0..7,16..23}/{8..15,24..31}
    FragBF fa[4];
    #pragma unroll
    for (int i = 0; i < 4; ++i) {
        const bf16_t* pa = cA + (wm + i * 16 + (lane & 15)) * LDA + ((lane >> 4) * 8);
        fa[i].q[0] = *(const u32x4*)pa;          // K base + {0..7}
        fa[i].q[1] = *(const u32x4*)(pa + 16);   // K base + {16..23}
    }
    #pragma unroll
    for (int i = 0; i < 4; ++i)
        #pragma unroll
        for (int j = 0; j < 2; ++j)
            acc[i][j] = __builtin_amdgcn_wmma_f32_16x16x32_bf16(
                false, fa[i].v, false, fb[j].v, (short)0, acc[i][j], false, false);
}

// ---------------------------------------------------------------------------
// GEMM: out[m,n] = act( sum_k A[m,k] * Wt[k,n] + bias[n] )
//   A  : [M, K] bf16 row-major
//   Wt : [K, N] bf16 row-major (pre-transposed weight)
//   EPI == 0 : out = tanh(.) stored bf16    (layer-1, feeds GEMM2)
//   EPI == 1 : out = (.)     stored f32     (delta)
// Block: 256 threads = 8 waves; tile 128x128; K-step 32.
// Wave (wm,wn) covers 64x32 = 4x2 WMMA 16x16 tiles.
// Double-buffered LDS filled by async Global->LDS DMA. Rolled 2-stage
// software-pipelined steady state (compact loop body, I$-friendly):
//   [issue(b1,k+32); wmma(b0,k); wait; bar; issue(b0,k+64); wmma(b1,k+32); wait; bar]*31
//   then: issue(b1,K-32); wmma(b0,K-64); wait; bar; wmma(b1,K-32)
// ---------------------------------------------------------------------------
template <int EPI>
__global__ __launch_bounds__(256) void vce_gemm_bf16(const bf16_t* __restrict__ A,
                                                     const bf16_t* __restrict__ Bt,
                                                     const float*  __restrict__ bias,
                                                     bf16_t* __restrict__ outB,
                                                     float*  __restrict__ outF) {
    constexpr int K = DIM, N = DIM;
    constexpr int LDA = 40;
    constexpr int LDB = 136;
    constexpr int ASZ = 128 * LDA;
    constexpr int BSZ = 32 * LDB;
    __shared__ bf16_t sA[2 * ASZ];
    __shared__ bf16_t sB[2 * BSZ];

    const int tid  = threadIdx.x;
    const int lane = tid & 31;
    const int wave = tid >> 5;
    const int mtile = blockIdx.y * 128;
    const int ntile = blockIdx.x * 128;
    const int wm = (wave & 1) * 64;     // wave M offset (2 waves in M)
    const int wn = (wave >> 1) * 32;    // wave N offset (4 waves in N)

    const v8f vzero = {0.f, 0.f, 0.f, 0.f, 0.f, 0.f, 0.f, 0.f};
    v8f acc[4][2];
    #pragma unroll
    for (int i = 0; i < 4; ++i)
        #pragma unroll
        for (int j = 0; j < 2; ++j) acc[i][j] = vzero;

    // cooperative DMA assignment: waves 0-3 fetch the A tile, waves 4-7 the B tile
    const bool loadsA = (tid < 128);
    const int  ar = tid;                 // A tile row 0..127
    const int  t2 = tid - 128;
    const int  bk = t2 >> 2;             // B tile k row 0..31
    const int  bq = t2 & 3;              // quarter of the 128-wide N row

    const bf16_t* gA = A  + (size_t)(mtile + ar) * K;               // + k0
    const bf16_t* gB = Bt + (size_t)bk * N + ntile + bq * 32;       // + k0*N
    const int ldsA_off = ar * LDA;                                  // 64B slice
    const int ldsB_off = bk * LDB + bq * 32;                        // 64B slice

    bf16_t* lA0 = sA + ldsA_off;        bf16_t* lA1 = lA0 + ASZ;
    bf16_t* lB0 = sB + ldsB_off;        bf16_t* lB1 = lB0 + BSZ;

    // prologue: stage tile k=0 into buffer 0
    if (loadsA) vce_async_row64(gA, lA0);
    else        vce_async_row64(gB, lB0);
    vce_wait_async0();
    __syncthreads();

    // steady state: rolled loop, two pipeline stages per iteration
    #pragma unroll 1
    for (int k0 = 0; k0 + 64 < K; k0 += 64) {
        // stage 0: compute buf0 (k0), DMA k0+32 -> buf1
        if (loadsA) vce_async_row64(gA + (k0 + 32), lA1);
        else        vce_async_row64(gB + (size_t)(k0 + 32) * N, lB1);
        vce_kstep(sA, sB, acc, lane, wm, wn);
        vce_wait_async0();
        __syncthreads();
        // stage 1: compute buf1 (k0+32), DMA k0+64 -> buf0
        if (loadsA) vce_async_row64(gA + (k0 + 64), lA0);
        else        vce_async_row64(gB + (size_t)(k0 + 64) * N, lB0);
        vce_kstep(sA + ASZ, sB + BSZ, acc, lane, wm, wn);
        vce_wait_async0();
        __syncthreads();
    }
    // tail: buf0 holds k=K-64; DMA k=K-32 -> buf1; compute both
    if (loadsA) vce_async_row64(gA + (K - 32), lA1);
    else        vce_async_row64(gB + (size_t)(K - 32) * N, lB1);
    vce_kstep(sA, sB, acc, lane, wm, wn);
    vce_wait_async0();
    __syncthreads();
    vce_kstep(sA + ASZ, sB + BSZ, acc, lane, wm, wn);

    // Epilogue. C layout: lanes 0-15 N=lane M=v, lanes 16-31 N=lane-16 M=8+v
    const int lhalf = lane >> 4, lcol = lane & 15;
    #pragma unroll
    for (int j = 0; j < 2; ++j) {
        const int ng = ntile + wn + j * 16 + lcol;
        const float bj = bias[ng];
        #pragma unroll
        for (int i = 0; i < 4; ++i) {
            const int mg = mtile + wm + i * 16 + lhalf * 8;
            #pragma unroll
            for (int v = 0; v < 8; ++v) {
                const float val = acc[i][j][v] + bj;
                if (EPI == 0) outB[(size_t)(mg + v) * N + ng] = (bf16_t)tanhf(val);
                else          outF[(size_t)(mg + v) * N + ng] = val;
            }
        }
    }
}

// ---------------------------------------------------------------------------
// Per-row vector-field update + norm cap. One 256-thread block per row.
// Uses ||h - dir||^2 = ||h||^2 - 2 h.dir + 1 (dirs are unit vectors).
// ---------------------------------------------------------------------------
__global__ __launch_bounds__(256) void vce_update(const float* __restrict__ h,
                                                  const float* __restrict__ delta,
                                                  const float* __restrict__ dirs,
                                                  float* __restrict__ out) {
    __shared__ float red[32];
    const int row  = blockIdx.x;
    const int tid  = threadIdx.x;
    const int lane = tid & 31, wave = tid >> 5;

    const float* hr = h     + (size_t)row * DIM;
    const float* dr = delta + (size_t)row * DIM;
    const float* e = dirs, *c = dirs + DIM, *n2 = dirs + 2 * DIM;

    float hv[8], ev[8], cv[8], nv[8];
    float s0 = 0.f, s1 = 0.f, s2 = 0.f, s3 = 0.f;
    #pragma unroll
    for (int t = 0; t < 8; ++t) {
        const int j = tid + t * 256;
        hv[t] = hr[j]; ev[t] = e[j]; cv[t] = c[j]; nv[t] = n2[j];
        s0 += hv[t] * hv[t];
        s1 += hv[t] * ev[t];
        s2 += hv[t] * cv[t];
        s3 += hv[t] * nv[t];
    }
    #pragma unroll
    for (int off = 16; off; off >>= 1) {
        s0 += __shfl_down(s0, off, 32);
        s1 += __shfl_down(s1, off, 32);
        s2 += __shfl_down(s2, off, 32);
        s3 += __shfl_down(s3, off, 32);
    }
    if (lane == 0) { red[wave] = s0; red[8 + wave] = s1; red[16 + wave] = s2; red[24 + wave] = s3; }
    __syncthreads();
    s0 = s1 = s2 = s3 = 0.f;
    #pragma unroll
    for (int w = 0; w < 8; ++w) {
        s0 += red[w]; s1 += red[8 + w]; s2 += red[16 + w]; s3 += red[24 + w];
    }
    __syncthreads();   // red[] free for reuse

    const float inv_nh = 1.f / fmaxf(sqrtf(s0), 1e-12f);
    const float de = 0.38f - s1 * inv_nh;
    const float dc = 0.38f - s2 * inv_nh;
    const float dn = 0.38f - s3 * inv_nh;
    const float ce = 0.10f * de / fmaxf(sqrtf(fmaxf(s0 - 2.f * s1 + 1.f, 0.f)), 1e-12f);
    const float cc = 0.10f * dc / fmaxf(sqrtf(fmaxf(s0 - 2.f * s2 + 1.f, 0.f)), 1e-12f);
    const float cn = 0.05f * dn / fmaxf(sqrtf(fmaxf(s0 - 2.f * s3 + 1.f, 0.f)), 1e-12f);

    float hnew[8];
    float sn = 0.f;
    #pragma unroll
    for (int t = 0; t < 8; ++t) {
        const int j = tid + t * 256;
        const float v = hv[t] + dr[j]
                      - ce * (hv[t] - ev[t])
                      - cc * (hv[t] - cv[t])
                      - cn * (hv[t] - nv[t]);
        hnew[t] = v;
        sn += v * v;
    }
    #pragma unroll
    for (int off = 16; off; off >>= 1) sn += __shfl_down(sn, off, 32);
    if (lane == 0) red[wave] = sn;
    __syncthreads();
    sn = 0.f;
    #pragma unroll
    for (int w = 0; w < 8; ++w) sn += red[w];

    const float nrm = sqrtf(sn);
    const float scl = (nrm > 10.f) ? (10.f / (nrm + 1e-8f)) : 1.f;
    #pragma unroll
    for (int t = 0; t < 8; ++t)
        out[(size_t)row * DIM + tid + t * 256] = hnew[t] * scl;
}

// ---------------------------------------------------------------------------
// Host-side launch
// ---------------------------------------------------------------------------
extern "C" void kernel_launch(void* const* d_in, const int* in_sizes, int n_in,
                              void* d_out, int out_size, void* d_ws, size_t ws_size,
                              hipStream_t stream) {
    const float* h0 = (const float*)d_in[0];
    const float* aE = (const float*)d_in[1];
    const float* aC = (const float*)d_in[2];
    const float* aN = (const float*)d_in[3];
    const float* W1 = (const float*)d_in[4];
    const float* b1 = (const float*)d_in[5];
    const float* W2 = (const float*)d_in[6];
    const float* b2 = (const float*)d_in[7];
    float* out = (float*)d_out;

    char* ws = (char*)d_ws;
    bf16_t* W1t  = (bf16_t*)ws;  ws += (size_t)DIM * DIM * sizeof(bf16_t);
    bf16_t* W2t  = (bf16_t*)ws;  ws += (size_t)DIM * DIM * sizeof(bf16_t);
    bf16_t* hbf  = (bf16_t*)ws;  ws += (size_t)BATCH * DIM * sizeof(bf16_t);
    bf16_t* tbf  = (bf16_t*)ws;  ws += (size_t)BATCH * DIM * sizeof(bf16_t);
    float*  delta = (float*)ws;  ws += (size_t)BATCH * DIM * sizeof(float);
    float*  dirs  = (float*)ws;  ws += (size_t)3 * DIM * sizeof(float);
    (void)ws_size; (void)in_sizes; (void)n_in; (void)out_size;

    vce_prep_dirs<<<1, 256, 0, stream>>>(aE, aC, aN, dirs);
    vce_wcvt<<<dim3(DIM / 32, DIM / 32), 256, 0, stream>>>(W1, W1t);
    vce_wcvt<<<dim3(DIM / 32, DIM / 32), 256, 0, stream>>>(W2, W2t);

    const dim3 ggrid(DIM / 128, BATCH / 128);   // (16, 128)
    const int  cvt_blocks = (BATCH * DIM) / (256 * 8);

    const float* hsrc = h0;
    for (int layer = 0; layer < 4; ++layer) {
        vce_cvt_h<<<cvt_blocks, 256, 0, stream>>>(hsrc, hbf);
        vce_gemm_bf16<0><<<ggrid, 256, 0, stream>>>(hbf, W1t, b1, tbf, nullptr);
        vce_gemm_bf16<1><<<ggrid, 256, 0, stream>>>(tbf, W2t, b2, nullptr, delta);
        vce_update<<<BATCH, 256, 0, stream>>>(hsrc, delta, dirs, out);
        hsrc = out;
    }
}
